// Attention_85186381348942
// MI455X (gfx1250) — matrix operand order repaired and checked by reading, not compiled, not run on hardware
//
#include <hip/hip_runtime.h>

#define HEADS 8
#define DHEAD 64
#define HID   512
#define NTOK  4096

typedef __bf16 bf16_t;
typedef __attribute__((ext_vector_type(16))) __bf16 v16bf;
typedef __attribute__((ext_vector_type(8)))  float  v8f;
typedef __attribute__((ext_vector_type(4)))  int    v4i;

union AFrag { v16bf v; unsigned int u[8]; };

#if defined(__has_builtin)
#  if __has_builtin(__builtin_amdgcn_global_load_async_to_lds_b128)
#    define HAVE_ASYNC_LDS 1
#  endif
#endif
#ifndef HAVE_ASYNC_LDS
#  define HAVE_ASYNC_LDS 0
#endif

// K-offset of the low element of VGPR-pair p in the 16-bit A/B matrix layouts
// (ISA 7.12.2; codegen-confirmed by ds_load_2addr offsets)
__device__ __forceinline__ int ab_base(int p, int half) {
  return (p < 4 ? 2 * p : 16 + 2 * (p - 4)) + 8 * half;
}

__device__ __forceinline__ v8f wmma_bf16(const AFrag& a, const AFrag& b, v8f c) {
  return __builtin_amdgcn_wmma_f32_16x16x32_bf16(false, a.v, false, b.v,
                                                 (short)0, c, false, false);
}

__device__ __forceinline__ unsigned pack_bf2(float x, float y) {
  union { unsigned u; __bf16 h[2]; } p;
  p.h[0] = (__bf16)x; p.h[1] = (__bf16)y;
  return p.u;
}

__device__ __forceinline__ void copy16_to_lds(const bf16_t* src, __bf16* dst) {
#if HAVE_ASYNC_LDS
  // signature (probe-confirmed): (v4i as1*, v4i as3*, imm off, imm cpol)
  __builtin_amdgcn_global_load_async_to_lds_b128(
      (__attribute__((address_space(1))) v4i*)(bf16_t*)src,
      (__attribute__((address_space(3))) v4i*)dst, 0, 0);
#else
  *(uint4*)dst = *(const uint4*)src;
#endif
}

__device__ __forceinline__ void wait_async_stage() {
#if HAVE_ASYNC_LDS
  asm volatile("s_wait_asynccnt 0x0" ::: "memory");
#endif
}

// ---------------------------------------------------------------------------
// Kernel 1: QKV projection. qkv[o,n] = sum_c W[o,c]*x[b,c,n], o in [0,1536).
// Q scaled by 1/8, layout [b][h][n][d]; K layout [b][h][n][d];
// V stored TRANSPOSED [b][h][d][n] so flash staging is a contiguous copy.
// Block = 256 thr = 8 waves, 64x128 tile, 2x2 C-frags per wave.
// ---------------------------------------------------------------------------
__global__ __launch_bounds__(256) void qkv_proj(const float* __restrict__ W,
                                                const float* __restrict__ X,
                                                bf16_t* __restrict__ Q,
                                                bf16_t* __restrict__ Kd,
                                                bf16_t* __restrict__ V) {
  __shared__ alignas(16) __bf16 Ash[64][34];    // [m][k]
  __shared__ alignas(16) __bf16 Bsh[128][34];   // [n][k] transposed

  const int tid  = threadIdx.x;
  const int wid  = tid >> 5, lane = tid & 31;
  const int half = lane >> 4, l16 = lane & 15;
  const int wm   = wid >> 2, wn = wid & 3;      // wave -> 32x32 sub-tile
  const int n0   = blockIdx.x * 128;
  const int m0   = blockIdx.y * 64;
  const int b    = blockIdx.z;
  const float* Xb = X + (size_t)b * HID * NTOK;

  v8f acc[2][2];
#pragma unroll
  for (int mi = 0; mi < 2; ++mi)
#pragma unroll
    for (int ni = 0; ni < 2; ++ni)
      acc[mi][ni] = (v8f){0.f, 0.f, 0.f, 0.f, 0.f, 0.f, 0.f, 0.f};

  for (int kc = 0; kc < HID; kc += 32) {
    __syncthreads();
    {
      // stage A: 64 rows x 32 k; all loads issued before converts
      int row = tid >> 2, c8 = (tid & 3) * 8;
      const float* asrc = W + (size_t)(m0 + row) * HID + kc + c8;
      const float4 a0 = *(const float4*)(asrc);
      const float4 a1 = *(const float4*)(asrc + 4);
      // stage B transposed: 32 k x 128 n; 4 independent b128 loads in flight
      int k = tid & 31, n16 = (tid >> 5) * 16;
      const float* src = Xb + (size_t)(kc + k) * NTOK + n0 + n16;
      const float4 b0 = *(const float4*)(src);
      const float4 b1 = *(const float4*)(src + 4);
      const float4 b2 = *(const float4*)(src + 8);
      const float4 b3 = *(const float4*)(src + 12);

      unsigned* d = (unsigned*)&Ash[row][c8];
      d[0] = pack_bf2(a0.x, a0.y); d[1] = pack_bf2(a0.z, a0.w);
      d[2] = pack_bf2(a1.x, a1.y); d[3] = pack_bf2(a1.z, a1.w);
      Bsh[n16 +  0][k] = (__bf16)b0.x; Bsh[n16 +  1][k] = (__bf16)b0.y;
      Bsh[n16 +  2][k] = (__bf16)b0.z; Bsh[n16 +  3][k] = (__bf16)b0.w;
      Bsh[n16 +  4][k] = (__bf16)b1.x; Bsh[n16 +  5][k] = (__bf16)b1.y;
      Bsh[n16 +  6][k] = (__bf16)b1.z; Bsh[n16 +  7][k] = (__bf16)b1.w;
      Bsh[n16 +  8][k] = (__bf16)b2.x; Bsh[n16 +  9][k] = (__bf16)b2.y;
      Bsh[n16 + 10][k] = (__bf16)b2.z; Bsh[n16 + 11][k] = (__bf16)b2.w;
      Bsh[n16 + 12][k] = (__bf16)b3.x; Bsh[n16 + 13][k] = (__bf16)b3.y;
      Bsh[n16 + 14][k] = (__bf16)b3.z; Bsh[n16 + 15][k] = (__bf16)b3.w;
    }
    __syncthreads();

    AFrag a[2], bb[2];
#pragma unroll
    for (int mi = 0; mi < 2; ++mi)
#pragma unroll
      for (int p = 0; p < 8; ++p)
        a[mi].u[p] = *(const unsigned*)&Ash[wm * 32 + mi * 16 + l16][ab_base(p, half)];
#pragma unroll
    for (int ni = 0; ni < 2; ++ni)
#pragma unroll
      for (int p = 0; p < 8; ++p)
        bb[ni].u[p] = *(const unsigned*)&Bsh[wn * 32 + ni * 16 + l16][ab_base(p, half)];
#pragma unroll
    for (int mi = 0; mi < 2; ++mi)
#pragma unroll
      for (int ni = 0; ni < 2; ++ni)
        acc[mi][ni] = wmma_bf16(a[mi], bb[ni], acc[mi][ni]);
  }

#pragma unroll
  for (int mi = 0; mi < 2; ++mi)
#pragma unroll
    for (int ni = 0; ni < 2; ++ni) {
      const int n = n0 + wn * 32 + ni * 16 + l16;
#pragma unroll
      for (int r = 0; r < 8; ++r) {
        int o   = m0 + wm * 32 + mi * 16 + r + 8 * half;
        int seg = o >> 9;                         // 0=Q 1=K 2=V
        int oo  = o & 511;
        int hh  = oo >> 6, dd = oo & 63;
        float vv = acc[mi][ni][r];
        if (seg == 0)
          Q[(((size_t)(b * HEADS + hh) * NTOK) + n) * DHEAD + dd] = (__bf16)(vv * 0.125f);
        else if (seg == 1)
          Kd[(((size_t)(b * HEADS + hh) * NTOK) + n) * DHEAD + dd] = (__bf16)vv;
        else  // V transposed: [b][h][d][n]
          V[(((size_t)(b * HEADS + hh) * DHEAD) + dd) * NTOK + n] = (__bf16)vv;
      }
    }
}

// ---------------------------------------------------------------------------
// Kernel 2: flash attention. Block = 8 waves x 32 query rows = 256 queries.
// DOUBLE-BUFFERED async global->LDS staging; the key loop is unrolled x2 so
// cur/nxt buffer bases are compile-time constants and the bodies are
// branch-free (last-iteration stage wraps to block 0: unused, memory-safe).
// ---------------------------------------------------------------------------
__global__ __launch_bounds__(256) void flash_attn(const bf16_t* __restrict__ Q,
                                                  const bf16_t* __restrict__ K,
                                                  const bf16_t* __restrict__ V,
                                                  bf16_t* __restrict__ Aout) {
  __shared__ alignas(16) __bf16 Ksh[2][32][72];  // [buf][key][d], 144B rows
  __shared__ alignas(16) __bf16 Vsh[2][64][40];  // [buf][d][key], 80B rows
  __shared__ alignas(16) __bf16 Psh[8][32][34];  // per-wave P scratch

  const int tid  = threadIdx.x;
  const int wid  = tid >> 5, lane = tid & 31;
  const int half = lane >> 4, l16 = lane & 15;
  const int h = blockIdx.y, b = blockIdx.z;

  const size_t hoff = (size_t)(b * HEADS + h) * NTOK * DHEAD;
  const bf16_t* Qh = Q + hoff;
  const bf16_t* Kh = K + hoff;
  const bf16_t* Vh = V + hoff;                    // transposed [d][n]
  const int q0 = blockIdx.x * 256 + wid * 32;

  const int kkey = tid >> 3, kch = (tid & 7) * 8; // K stage: 16B/thread
  const int vd   = tid >> 2, vch = (tid & 3) * 8; // V stage: 16B/thread

  auto stage = [&](int buf, int j0) {
    copy16_to_lds(Kh + (size_t)(j0 + kkey) * DHEAD + kch, &Ksh[buf][kkey][kch]);
    copy16_to_lds(Vh + (size_t)vd * NTOK + j0 + vch, &Vsh[buf][vd][vch]);
  };

  // Q A-frags: 2 query tiles x 2 K-dim chunks, loaded once
  AFrag qa[2][2];
#pragma unroll
  for (int qt = 0; qt < 2; ++qt)
#pragma unroll
    for (int kb = 0; kb < 2; ++kb)
#pragma unroll
      for (int p = 0; p < 8; ++p)
        qa[qt][kb].u[p] = *(const unsigned*)(Qh + (size_t)(q0 + qt * 16 + l16) * DHEAD +
                                             kb * 32 + ab_base(p, half));

  v8f oacc[2][4];
#pragma unroll
  for (int qt = 0; qt < 2; ++qt)
#pragma unroll
    for (int dt = 0; dt < 4; ++dt)
      oacc[qt][dt] = (v8f){0.f, 0.f, 0.f, 0.f, 0.f, 0.f, 0.f, 0.f};
  float mrow[2][8], lrow[2][8];
#pragma unroll
  for (int qt = 0; qt < 2; ++qt)
#pragma unroll
    for (int r = 0; r < 8; ++r) { mrow[qt][r] = -1e30f; lrow[qt][r] = 0.f; }

  // prologue: stage first tile, wait for it
  stage(0, 0);
  wait_async_stage();
  __syncthreads();

#pragma unroll 2
  for (int j0 = 0; j0 < NTOK; j0 += 32) {
    const int cur = (j0 >> 5) & 1, nxt = cur ^ 1;

    // kick off async DMA for the NEXT key block into the other buffer
    // (wraps to block 0 on the final iteration: unused but memory-safe)
    stage(nxt, (j0 + 32) & (NTOK - 1));
    // warm L2 two blocks ahead
    __builtin_prefetch(Kh + (size_t)((j0 + 64) & (NTOK - 1)) * DHEAD + kch, 0, 1);

    // K and V B-frags from the CURRENT buffer, shared across both query tiles
    AFrag kf[2][2];
#pragma unroll
    for (int sub = 0; sub < 2; ++sub)
#pragma unroll
      for (int kb = 0; kb < 2; ++kb)
#pragma unroll
        for (int p = 0; p < 8; ++p)
          kf[sub][kb].u[p] = *(const unsigned*)
              &Ksh[cur][sub * 16 + l16][kb * 32 + ab_base(p, half)];
    AFrag vb[4];
#pragma unroll
    for (int dt = 0; dt < 4; ++dt)
#pragma unroll
      for (int p = 0; p < 8; ++p)
        vb[dt].u[p] = *(const unsigned*)&Vsh[cur][dt * 16 + l16][ab_base(p, half)];

#pragma unroll
    for (int qt = 0; qt < 2; ++qt) {
      v8f s[2];
      s[0] = (v8f){0.f, 0.f, 0.f, 0.f, 0.f, 0.f, 0.f, 0.f};
      s[1] = s[0];
#pragma unroll
      for (int sub = 0; sub < 2; ++sub)
#pragma unroll
        for (int kb = 0; kb < 2; ++kb)
          s[sub] = wmma_bf16(qa[qt][kb], kf[sub][kb], s[sub]);

      // online softmax (C-frag rows live on 16-lane groups)
#pragma unroll
      for (int r = 0; r < 8; ++r) {
        float mb = fmaxf(s[0][r], s[1][r]);
#pragma unroll
        for (int msk = 8; msk >= 1; msk >>= 1)
          mb = fmaxf(mb, __shfl_xor(mb, msk, 32));
        float mn    = fmaxf(mrow[qt][r], mb);
        float alpha = __expf(mrow[qt][r] - mn);
        mrow[qt][r] = mn;
        float p0 = __expf(s[0][r] - mn);
        float p1 = __expf(s[1][r] - mn);
        s[0][r] = p0; s[1][r] = p1;
        float rs = p0 + p1;
#pragma unroll
        for (int msk = 8; msk >= 1; msk >>= 1)
          rs += __shfl_xor(rs, msk, 32);
        lrow[qt][r] = lrow[qt][r] * alpha + rs;
#pragma unroll
        for (int dt = 0; dt < 4; ++dt) oacc[qt][dt][r] *= alpha;
      }

      // C-layout P -> LDS (A-layout reload below)
#pragma unroll
      for (int sub = 0; sub < 2; ++sub)
#pragma unroll
        for (int r = 0; r < 8; ++r)
          Psh[wid][qt * 16 + r + 8 * half][sub * 16 + l16] = (__bf16)s[sub][r];
    }
    asm volatile("s_wait_dscnt 0x0" ::: "memory");  // intra-wave LDS RAW (P only)

    AFrag pa[2];
#pragma unroll
    for (int qt = 0; qt < 2; ++qt)
#pragma unroll
      for (int p = 0; p < 8; ++p)
        pa[qt].u[p] = *(const unsigned*)&Psh[wid][qt * 16 + l16][ab_base(p, half)];

#pragma unroll
    for (int qt = 0; qt < 2; ++qt)
#pragma unroll
      for (int dt = 0; dt < 4; ++dt)
        oacc[qt][dt] = wmma_bf16(pa[qt], vb[dt], oacc[qt][dt]);

    // next tile must have landed before anyone reads it; cur is free to reuse
    wait_async_stage();
    __syncthreads();
  }

  // normalize, write attention output as [b][c = h*64+d][n] bf16
  bf16_t* dst = Aout + ((size_t)b * HID + h * DHEAD) * NTOK;
#pragma unroll
  for (int qt = 0; qt < 2; ++qt)
#pragma unroll
    for (int r = 0; r < 8; ++r) {
      float inv = 1.f / lrow[qt][r];
      int n = q0 + qt * 16 + r + 8 * half;
#pragma unroll
      for (int dt = 0; dt < 4; ++dt) {
        int d = dt * 16 + l16;
        dst[(size_t)d * NTOK + n] = (__bf16)(oacc[qt][dt][r] * inv);
      }
    }
}

// ---------------------------------------------------------------------------
// Kernel 3: output projection + bias (bf16 B operand).
// ---------------------------------------------------------------------------
__global__ __launch_bounds__(256) void out_proj(const float* __restrict__ W,
                                                const bf16_t* __restrict__ Ain,
                                                const float* __restrict__ bias,
                                                float* __restrict__ Out) {
  __shared__ alignas(16) __bf16 Ash[64][34];
  __shared__ alignas(16) __bf16 Bsh[128][34];

  const int tid  = threadIdx.x;
  const int wid  = tid >> 5, lane = tid & 31;
  const int half = lane >> 4, l16 = lane & 15;
  const int wm   = wid >> 2, wn = wid & 3;
  const int n0   = blockIdx.x * 128;
  const int m0   = blockIdx.y * 64;
  const int b    = blockIdx.z;

  v8f acc[2][2];
#pragma unroll
  for (int mi = 0; mi < 2; ++mi)
#pragma unroll
    for (int ni = 0; ni < 2; ++ni)
      acc[mi][ni] = (v8f){0.f, 0.f, 0.f, 0.f, 0.f, 0.f, 0.f, 0.f};

  for (int kc = 0; kc < HID; kc += 32) {
    __syncthreads();
    {
      int row = tid >> 2, c8 = (tid & 3) * 8;
      const float* asrc = W + (size_t)(m0 + row) * HID + kc + c8;
      const float4 a0 = *(const float4*)(asrc);
      const float4 a1 = *(const float4*)(asrc + 4);
      int k = tid & 31, n16 = (tid >> 5) * 16;
      const bf16_t* src = Ain + ((size_t)b * HID + kc + k) * NTOK + n0 + n16;
      uint4 r0 = *(const uint4*)(src);
      uint4 r1 = *(const uint4*)(src + 8);

      unsigned* d = (unsigned*)&Ash[row][c8];
      d[0] = pack_bf2(a0.x, a0.y); d[1] = pack_bf2(a0.z, a0.w);
      d[2] = pack_bf2(a1.x, a1.y); d[3] = pack_bf2(a1.z, a1.w);
      const __bf16* hv0 = (const __bf16*)&r0;
      const __bf16* hv1 = (const __bf16*)&r1;
#pragma unroll
      for (int i = 0; i < 8; ++i) Bsh[n16 + i][k] = hv0[i];
#pragma unroll
      for (int i = 0; i < 8; ++i) Bsh[n16 + 8 + i][k] = hv1[i];
    }
    __syncthreads();

    AFrag a[2], bb[2];
#pragma unroll
    for (int mi = 0; mi < 2; ++mi)
#pragma unroll
      for (int p = 0; p < 8; ++p)
        a[mi].u[p] = *(const unsigned*)&Ash[wm * 32 + mi * 16 + l16][ab_base(p, half)];
#pragma unroll
    for (int ni = 0; ni < 2; ++ni)
#pragma unroll
      for (int p = 0; p < 8; ++p)
        bb[ni].u[p] = *(const unsigned*)&Bsh[wn * 32 + ni * 16 + l16][ab_base(p, half)];
#pragma unroll
    for (int mi = 0; mi < 2; ++mi)
#pragma unroll
      for (int ni = 0; ni < 2; ++ni)
        acc[mi][ni] = wmma_bf16(a[mi], bb[ni], acc[mi][ni]);
  }

#pragma unroll
  for (int mi = 0; mi < 2; ++mi)
#pragma unroll
    for (int ni = 0; ni < 2; ++ni) {
      const int n = n0 + wn * 32 + ni * 16 + l16;
#pragma unroll
      for (int r = 0; r < 8; ++r) {
        int o = m0 + wm * 32 + mi * 16 + r + 8 * half;
        Out[((size_t)b * HID + o) * NTOK + n] = acc[mi][ni][r] + bias[o];
      }
    }
}

// ---------------------------------------------------------------------------
extern "C" void kernel_launch(void* const* d_in, const int* in_sizes, int n_in,
                              void* d_out, int out_size, void* d_ws, size_t ws_size,
                              hipStream_t stream) {
  const float* x     = (const float*)d_in[0];
  const float* w_qkv = (const float*)d_in[1];
  const float* w_out = (const float*)d_in[2];
  const float* b_out = (const float*)d_in[3];
  float* out = (float*)d_out;

  const size_t per = (size_t)2 * HEADS * NTOK * DHEAD;  // 4M bf16 elems each
  bf16_t* Q    = (bf16_t*)d_ws;
  bf16_t* K    = Q + per;
  bf16_t* V    = K + per;     // transposed [b][h][d][n]
  bf16_t* aout = V + per;     // [b][512][4096] bf16

  qkv_proj<<<dim3(NTOK / 128, (3 * HID) / 64, 2), 256, 0, stream>>>(w_qkv, x, Q, K, V);
  flash_attn<<<dim3(NTOK / 256, HEADS, 2), 256, 0, stream>>>(Q, K, V, aout);
  out_proj<<<dim3(NTOK / 128, HID / 64, 2), 256, 0, stream>>>(w_out, aout, b_out, out);
}
